// TSMNetWithGradientFlow_22643067584709
// MI455X (gfx1250) — matrix operand order repaired and checked by reading, not compiled, not run on hardware
//
#include <hip/hip_runtime.h>
#include <hip/hip_bf16.h>

typedef __attribute__((ext_vector_type(16))) _Float16 v16h;
typedef __attribute__((ext_vector_type(8)))  _Float16 v8h;
typedef __attribute__((ext_vector_type(8)))  float    v8f;

#define NB   128     // batch
#define NC   64      // channels
#define NT   4000    // time
#define NF1  4
#define NK   25
#define NPAD 12
#define NF2  40
#define NP   20
#define TT   32      // time tile
#define KKR  256     // F1*C reduction

// ---------------------------------------------------------------------------
// Kernel 1: fused conv1 + conv2 (WMMA) + covariance (WMMA) + BiMap -> S[b]
// ---------------------------------------------------------------------------

__device__ __forceinline__ int reflect_idx(int t) {
  if (t < 0) t = -t;
  if (t >= NT) t = 2 * NT - 2 - t;
  return t;
}

struct CovSmem {
  float xs[64][57];                 // x tile (C x (TT+24)), f32
  _Float16 w2h[48][256];            // conv2 weights, M padded to 48 (rows>=40 zero)
  union {
    _Float16 h1T[32][264];          // h1 transposed [tt][kk], padded pitch (16B aligned)
    float    Gf[48][48];            // reused after T loop: raw Gram matrix
  };
  union {
    _Float16 h2s[48][40];           // h2 tile [f2][tt] (rows>=40 zero), f16
    float    tmpPG[20][40];         // reused: bimap * Cov
  };
  float w1s[4][25];
  float b1s[4];
  float b2s[40];
  float msum[40];
};

__global__ __launch_bounds__(192, 1) void cov_bimap_kernel(
    const float* __restrict__ x,  const float* __restrict__ w1,
    const float* __restrict__ b1, const float* __restrict__ w2,
    const float* __restrict__ b2, const float* __restrict__ bimap,
    float* __restrict__ Sout) {
  __shared__ CovSmem sm;
  const int b   = blockIdx.x;
  const int tid = threadIdx.x;
  const int w   = tid >> 5;       // wave id 0..5 (wave32)
  const int l   = tid & 31;       // lane id

  // ---- preload weights into LDS -------------------------------------------
  for (int e = tid; e < 48 * 256; e += 192) {
    int m = e >> 8, kk = e & 255;
    sm.w2h[m][kk] = (m < NF2) ? (_Float16)w2[m * 256 + kk] : (_Float16)0.f;
  }
  for (int e = tid; e < 100; e += 192) sm.w1s[e / 25][e % 25] = w1[e];
  if (tid < 4)  sm.b1s[tid] = b1[tid];
  if (tid < 40) { sm.b2s[tid] = b2[tid]; sm.msum[tid] = 0.f; }

  // cov tile ownership: 6 upper-triangle 16x16 tiles of the 48x48 Gram
  const int pm[6] = {0, 0, 0, 1, 1, 2};
  const int pn[6] = {0, 1, 2, 1, 2, 2};
  const int mi = pm[w], ni = pn[w];
  const int mC = w % 3, nC = w / 3;          // conv2 output tile (m,n)
  const int khalf = l >> 4;
  v8f ccov = {0.f, 0.f, 0.f, 0.f, 0.f, 0.f, 0.f, 0.f};
  __syncthreads();

  for (int t0 = 0; t0 < NT; t0 += TT) {
    // ---- stage A: load x tile (with reflect padding) ----------------------
    for (int e = tid; e < 64 * 56; e += 192) {
      int c = e / 56, j = e % 56;
      sm.xs[c][j] = x[((long)b * NC + c) * NT + reflect_idx(t0 + j - NPAD)];
    }
    __syncthreads();

    // ---- stage B: conv1 (25-tap FIR) -> h1T f16 ---------------------------
    for (int e = tid; e < KKR * TT; e += 192) {
      int kk = e >> 5, tt = e & 31;
      int f1 = kk >> 6, c = kk & 63;
      float acc = sm.b1s[f1];
#pragma unroll
      for (int k = 0; k < NK; ++k) acc += sm.w1s[f1][k] * sm.xs[c][tt + k];
      sm.h1T[tt][kk] = (_Float16)acc;
    }
    __syncthreads();

    // ---- stage C: conv2 = W2(48x256) x h1(256x32) via WMMA ----------------
    {
      v8f acc = {0.f, 0.f, 0.f, 0.f, 0.f, 0.f, 0.f, 0.f};
      const int rowA = mC * 16 + (l & 15);
      const int colB = nC * 16 + (l & 15);
#pragma unroll
      for (int ks = 0; ks < 8; ++ks) {
        int baseA = ks * 32 + khalf * 8;
        v8h alo = *(const v8h*)&sm.w2h[rowA][baseA];
        v8h ahi = *(const v8h*)&sm.w2h[rowA][baseA + 16];
        v16h afr = __builtin_shufflevector(alo, ahi, 0,1,2,3,4,5,6,7,8,9,10,11,12,13,14,15);
        int baseB = ks * 32 + khalf * 8;
        v8h blo = *(const v8h*)&sm.h1T[colB][baseB];
        v8h bhi = *(const v8h*)&sm.h1T[colB][baseB + 16];
        v16h bfr = __builtin_shufflevector(blo, bhi, 0,1,2,3,4,5,6,7,8,9,10,11,12,13,14,15);
        acc = __builtin_amdgcn_wmma_f32_16x16x32_f16(false, afr, false, bfr,
                                                     (short)0, acc, false, false);
      }
      int Mbase = mC * 16 + (khalf << 3);
#pragma unroll
      for (int r = 0; r < 8; ++r) {
        int Mr = Mbase + r;
        float v = (Mr < NF2) ? (acc[r] + sm.b2s[Mr]) : 0.f;
        sm.h2s[Mr][colB] = (_Float16)v;
      }
    }
    __syncthreads();

    // ---- stage D: covariance accumulate H * H^T (WMMA) + row sums ---------
    {
      int rowA = mi * 16 + (l & 15);
      v8h alo = *(const v8h*)&sm.h2s[rowA][khalf * 8];
      v8h ahi = *(const v8h*)&sm.h2s[rowA][16 + khalf * 8];
      v16h afr = __builtin_shufflevector(alo, ahi, 0,1,2,3,4,5,6,7,8,9,10,11,12,13,14,15);
      int rowB = ni * 16 + (l & 15);
      v8h blo = *(const v8h*)&sm.h2s[rowB][khalf * 8];
      v8h bhi = *(const v8h*)&sm.h2s[rowB][16 + khalf * 8];
      v16h bfr = __builtin_shufflevector(blo, bhi, 0,1,2,3,4,5,6,7,8,9,10,11,12,13,14,15);
      ccov = __builtin_amdgcn_wmma_f32_16x16x32_f16(false, afr, false, bfr,
                                                    (short)0, ccov, false, false);
    }
    if (tid < 40) {
      float s = 0.f;
      for (int tt = 0; tt < TT; ++tt) s += (float)sm.h2s[tid][tt];
      sm.msum[tid] += s;
    }
    __syncthreads();
  }

  // ---- write cov tiles (and mirrors) into Gf (aliases h1T) ----------------
  {
    int Nc = ni * 16 + (l & 15);
    int Mb = mi * 16 + (khalf << 3);
#pragma unroll
    for (int r = 0; r < 8; ++r) {
      int Mr = Mb + r;
      float v = ccov[r];
      sm.Gf[Mr][Nc] = v;
      sm.Gf[Nc][Mr] = v;
    }
  }
  __syncthreads();

  // ---- finalize covariance: (G - T*m*m^T)/(T-1), f,g < 40 -----------------
  for (int e = tid; e < 1600; e += 192) {
    int f = e / 40, g = e % 40;
    sm.Gf[f][g] = (sm.Gf[f][g] - sm.msum[f] * sm.msum[g] * (1.f / (float)NT))
                  * (1.f / (float)(NT - 1));
  }
  __syncthreads();

  // ---- BiMap: tmpPG = Wb(20x40) * Cov(40x40) ------------------------------
  for (int e = tid; e < 800; e += 192) {
    int p = e / 40, g = e % 40;
    float a = 0.f;
    for (int f = 0; f < 40; ++f) a += bimap[p * 40 + f] * sm.Gf[f][g];
    sm.tmpPG[p][g] = a;
  }
  __syncthreads();

  // ---- S = 0.5 * (Wb C Wb^T + (Wb C Wb^T)^T) ------------------------------
  for (int e = tid; e < 400; e += 192) {
    int p = e / 20, q = e % 20;
    float a = 0.f, bs = 0.f;
    for (int g = 0; g < 40; ++g) {
      a  += sm.tmpPG[p][g] * bimap[q * 40 + g];
      bs += sm.tmpPG[q][g] * bimap[p * 40 + g];
    }
    Sout[b * 400 + e] = 0.5f * (a + bs);
  }
}

// ---------------------------------------------------------------------------
// 20x20 symmetric eigensolver (two-sided cyclic Jacobi), block-cooperative
// ---------------------------------------------------------------------------
__device__ void jacobi20(float (*A)[21], float (*V)[21]) {
  const int tid = threadIdx.x;
  __shared__ float jcs[2];
  for (int e = tid; e < 400; e += blockDim.x)
    V[e / 20][e % 20] = (e / 20 == e % 20) ? 1.f : 0.f;
  __syncthreads();
  for (int sweep = 0; sweep < 15; ++sweep) {
    for (int p = 0; p < 19; ++p) {
      for (int q = p + 1; q < 20; ++q) {
        if (tid == 0) {
          float apq = A[p][q];
          float th = (apq == 0.f) ? 0.f : 0.5f * atan2f(2.f * apq, A[q][q] - A[p][p]);
          jcs[0] = cosf(th);
          jcs[1] = sinf(th);
        }
        __syncthreads();
        float c = jcs[0], s = jcs[1];
        if (tid < 20) {                        // rows p,q
          float ap = A[p][tid], aq = A[q][tid];
          A[p][tid] = c * ap - s * aq;
          A[q][tid] = s * ap + c * aq;
        }
        __syncthreads();
        if (tid < 20) {                        // cols p,q (+ eigenvectors)
          float ap = A[tid][p], aq = A[tid][q];
          A[tid][p] = c * ap - s * aq;
          A[tid][q] = s * ap + c * aq;
          float vp = V[tid][p], vq = V[tid][q];
          V[tid][p] = c * vp - s * vq;
          V[tid][q] = s * vp + c * vq;
        }
        __syncthreads();
      }
    }
  }
}

#define MODE_CLAMP 0
#define MODE_SQRT  1
#define MODE_RSQRT 2
#define MODE_LOG   3
#define MODE_EXP   4
#define MODE_SQRT_BOTH 5

// out = U f(s) U^T of in[b]; mode 5 writes sqrt->out and rsqrt->out2
__global__ void sym_fn_kernel(const float* __restrict__ in, float* __restrict__ out,
                              float* __restrict__ out2, int mode) {
  __shared__ float A_[20][21], V_[20][21], fl1[20], fl2[20];
  const int b = blockIdx.x;
  const int tid = threadIdx.x;
  for (int e = tid; e < 400; e += blockDim.x) A_[e / 20][e % 20] = in[b * 400 + e];
  __syncthreads();
  jacobi20(A_, V_);
  if (tid < 20) {
    float lam = A_[tid][tid];
    float f = 0.f, f2 = 0.f;
    switch (mode) {
      case MODE_CLAMP: f = fmaxf(lam, 1e-4f); break;
      case MODE_SQRT:  f = sqrtf(fmaxf(lam, 1e-30f)); break;
      case MODE_RSQRT: f = rsqrtf(fmaxf(lam, 1e-30f)); break;
      case MODE_LOG:   f = logf(fmaxf(lam, 1e-30f)); break;
      case MODE_EXP:   f = expf(lam); break;
      default: { float cl = fmaxf(lam, 1e-30f); f = sqrtf(cl); f2 = rsqrtf(cl); } break;
    }
    fl1[tid] = f; fl2[tid] = f2;
  }
  __syncthreads();
  for (int e = tid; e < 400; e += blockDim.x) {
    int i = e / 20, j = e % 20;
    float a = 0.f, a2 = 0.f;
    for (int k = 0; k < 20; ++k) {
      float vv = V_[i][k] * V_[j][k];
      a  += fl1[k] * vv;
      a2 += fl2[k] * vv;
    }
    out[b * 400 + e] = a;
    if (out2) out2[b * 400 + e] = a2;
  }
}

// out = scale * sum_b in[b]
__global__ void mean_kernel(const float* __restrict__ in, float* __restrict__ out,
                            int n, float scale) {
  for (int e = threadIdx.x; e < 400; e += blockDim.x) {
    float a = 0.f;
    for (int b = 0; b < n; ++b) a += in[b * 400 + e];
    out[e] = a * scale;
  }
}

// per b: logm( sym( Gis * S_b * Gis ) ) -> out[b]; optional sum(L^2) -> var_acc
__global__ void whiten_log_kernel(const float* __restrict__ S, const float* __restrict__ Gis,
                                  float* __restrict__ out, float* __restrict__ var_acc) {
  __shared__ float W[400], X[400], T1[400];
  __shared__ float A_[20][21], V_[20][21], fl[20], red[64];
  const int b = blockIdx.x;
  const int tid = threadIdx.x;
  for (int e = tid; e < 400; e += blockDim.x) { W[e] = Gis[e]; X[e] = S[b * 400 + e]; }
  __syncthreads();
  for (int e = tid; e < 400; e += blockDim.x) {
    int i = e / 20, j = e % 20;
    float a = 0.f;
    for (int k = 0; k < 20; ++k) a += W[i * 20 + k] * X[k * 20 + j];
    T1[e] = a;
  }
  __syncthreads();
  for (int e = tid; e < 400; e += blockDim.x) {
    int i = e / 20, j = e % 20;
    float a = 0.f, a2 = 0.f;
    for (int k = 0; k < 20; ++k) {
      a  += T1[i * 20 + k] * W[k * 20 + j];
      a2 += T1[j * 20 + k] * W[k * 20 + i];
    }
    A_[i][j] = 0.5f * (a + a2);
  }
  __syncthreads();
  jacobi20(A_, V_);
  if (tid < 20) fl[tid] = logf(fmaxf(A_[tid][tid], 1e-30f));
  __syncthreads();
  float psum = 0.f;
  for (int e = tid; e < 400; e += blockDim.x) {
    int i = e / 20, j = e % 20;
    float a = 0.f;
    for (int k = 0; k < 20; ++k) a += V_[i][k] * fl[k] * V_[j][k];
    out[b * 400 + e] = a;
    psum += a * a;
  }
  if (var_acc) {
    red[tid] = psum;
    __syncthreads();
    if (tid == 0) {
      float t = 0.f;
      for (int i = 0; i < (int)blockDim.x; ++i) t += red[i];
      atomicAdd(var_acc, t);
    }
  }
}

// G = Gs * E * Gs
__global__ void gupdate_kernel(const float* __restrict__ Gsq, const float* __restrict__ E,
                               float* __restrict__ Gout) {
  __shared__ float A1[400], B1[400], T1[400];
  const int tid = threadIdx.x;
  for (int e = tid; e < 400; e += blockDim.x) { A1[e] = Gsq[e]; B1[e] = E[e]; }
  __syncthreads();
  for (int e = tid; e < 400; e += blockDim.x) {
    int i = e / 20, j = e % 20;
    float a = 0.f;
    for (int k = 0; k < 20; ++k) a += A1[i * 20 + k] * B1[k * 20 + j];
    T1[e] = a;
  }
  __syncthreads();
  for (int e = tid; e < 400; e += blockDim.x) {
    int i = e / 20, j = e % 20;
    float a = 0.f;
    for (int k = 0; k < 20; ++k) a += T1[i * 20 + k] * A1[k * 20 + j];
    Gout[e] = a;
  }
}

// out[b][cls] = sum_triu( p * Lw * scale * head_w ) + head_b
__global__ void head_kernel(const float* __restrict__ Lw, const float* __restrict__ var_acc,
                            const float* __restrict__ bnw, const float* __restrict__ hw,
                            const float* __restrict__ hb, float* __restrict__ out) {
  __shared__ float L[400];
  const int b = blockIdx.x;
  const int tid = threadIdx.x;
  for (int e = tid; e < 400; e += blockDim.x) L[e] = Lw[b * 400 + e];
  __syncthreads();
  float p = bnw[0] * rsqrtf(var_acc[0] * (1.f / (float)NB) + 1e-5f);
  const float r2 = 1.41421356237309515f;
  if (tid < 4) {
    float acc = hb[tid];
    int e = 0;
    for (int i = 0; i < 20; ++i)
      for (int j = i; j < 20; ++j, ++e) {
        float sc = (i == j) ? 1.f : r2;
        acc += p * L[i * 20 + j] * sc * hw[tid * 210 + e];
      }
    out[b * 4 + tid] = acc;
  }
}

// ---------------------------------------------------------------------------
extern "C" void kernel_launch(void* const* d_in, const int* in_sizes, int n_in,
                              void* d_out, int out_size, void* d_ws, size_t ws_size,
                              hipStream_t stream) {
  (void)in_sizes; (void)n_in; (void)out_size; (void)ws_size;
  const float* x     = (const float*)d_in[0];
  const float* w1    = (const float*)d_in[1];
  const float* b1    = (const float*)d_in[2];
  const float* w2    = (const float*)d_in[3];
  const float* b2    = (const float*)d_in[4];
  const float* bimap = (const float*)d_in[5];
  const float* bnw   = (const float*)d_in[6];
  const float* hw    = (const float*)d_in[7];
  const float* hb    = (const float*)d_in[8];
  float* out = (float*)d_out;

  float* ws   = (float*)d_ws;
  float* S    = ws;                 // 128*400
  float* L    = S + NB * 400;       // 128*400
  float* G    = L + NB * 400;       // 400
  float* Gsq  = G + 400;            // 400
  float* Gis  = Gsq + 400;          // 400
  float* Lbar = Gis + 400;          // 400
  float* Ebuf = Lbar + 400;         // 400
  float* var  = Ebuf + 400;         // 1

  // S[b] = sym(Wb Cov_b Wb^T)
  cov_bimap_kernel<<<NB, 192, 0, stream>>>(x, w1, b1, w2, b2, bimap, S);
  // ReEig: clamp eigenvalues
  sym_fn_kernel<<<NB, 32, 0, stream>>>(S, S, nullptr, MODE_CLAMP);
  // G = mean(S)
  mean_kernel<<<1, 256, 0, stream>>>(S, G, NB, 1.f / (float)NB);
  // Karcher step (1x): Gs, Gis; Lb = log(Gis S Gis); G = Gs exp(mean Lb) Gs
  sym_fn_kernel<<<1, 32, 0, stream>>>(G, Gsq, Gis, MODE_SQRT_BOTH);
  whiten_log_kernel<<<NB, 64, 0, stream>>>(S, Gis, L, nullptr);
  mean_kernel<<<1, 256, 0, stream>>>(L, Lbar, NB, 1.f / (float)NB);
  sym_fn_kernel<<<1, 32, 0, stream>>>(Lbar, Ebuf, nullptr, MODE_EXP);
  gupdate_kernel<<<1, 64, 0, stream>>>(Gsq, Ebuf, G);
  // final whitening + log-map + variance
  sym_fn_kernel<<<1, 32, 0, stream>>>(G, Gis, nullptr, MODE_RSQRT);
  hipMemsetAsync(var, 0, sizeof(float), stream);
  whiten_log_kernel<<<NB, 64, 0, stream>>>(S, Gis, L, var);
  // head
  head_kernel<<<NB, 32, 0, stream>>>(L, var, bnw, hw, hb, out);
}
